// MTRNNCell_4715874091405
// MI455X (gfx1250) — matrix operand order repaired and checked by reading, not compiled, not run on hardware
//
#include <hip/hip_runtime.h>
#include <cmath>

typedef __attribute__((ext_vector_type(16))) __bf16 v16bf;
typedef __attribute__((ext_vector_type(8)))  float  v8f;

#define BM 128
#define BN 64
#define BK 32
#define LDK 40   // padded K stride (halves): 80 bytes/row, 16B-aligned

// pack two floats into two bf16 (round-half-up) with one v_perm_b32
static __device__ __forceinline__ unsigned pack2bf(float a, float b) {
    unsigned ua = __float_as_uint(a) + 0x8000u;
    unsigned ub = __float_as_uint(b) + 0x8000u;
    // bytes: d0=ua.b2, d1=ua.b3, d2=ub.b2, d3=ub.b3  ({ub,ua} concat, sel 0..3 -> ua)
    return __builtin_amdgcn_perm(ub, ua, 0x07060302u);
}

// single float -> bf16 (round-half-up); >>16 lets compiler use ds_store_b16_d16_hi
static __device__ __forceinline__ __bf16 f2bf(float f) {
    unsigned u = __float_as_uint(f) + 0x8000u;
    unsigned short hs = (unsigned short)(u >> 16);
    __bf16 h;
    __builtin_memcpy(&h, &hs, 2);
    return h;
}

struct FusedArgs {
    const float* A0; const float* W0; const float* b0; int K0;
    const float* A1; const float* W1; const float* b1; int K1;
    const float* A2; const float* W2; const float* b2; int K2;
    int nseg;
    const float* prev;   // previous state for leak (nullptr => pure tanh(sum+bias))
    float* out;
    int N;
    float tau;
};

static __device__ __forceinline__ void run_problem(const FusedArgs& args, int ntile) {
    __shared__ __align__(16) __bf16 As[BM][LDK];
    __shared__ __align__(16) __bf16 Bt[BN][LDK];   // transposed: [n][k]
    __shared__ float bsum[BN];

    const int tid  = threadIdx.x;
    const int lane = tid & 31;
    const int w    = tid >> 5;
    const int wm   = w & 3;     // 4 wave-rows of 32 M each
    const int wn   = w >> 2;    // 2 wave-cols of 32 N each
    const int lm   = lane & 15;
    const int lhi  = lane >> 4;

    const int       blockN = ntile * BN;
    const long long blockM = (long long)blockIdx.y * BM;
    const int       N      = args.N;

    // combined bias (each _lin in a segment adds its own bias; they sum)
    if (tid < BN) {
        float s = args.b0[blockN + tid];
        if (args.nseg > 1) s += args.b1[blockN + tid];
        if (args.nseg > 2) s += args.b2[blockN + tid];
        bsum[tid] = s;
    }

    v8f acc00 = {}, acc01 = {}, acc10 = {}, acc11 = {};

    const float* Aseg[3] = {args.A0, args.A1, args.A2};
    const float* Wseg[3] = {args.W0, args.W1, args.W2};
    const int    Kseg[3] = {args.K0, args.K1, args.K2};

    for (int s = 0; s < args.nseg; ++s) {
        const float* Ag = Aseg[s];
        const float* Wg = Wseg[s];
        const int    K  = Kseg[s];

        for (int kg = 0; kg < K; kg += BK) {
            __syncthreads();   // protect previous iteration's fragment reads

            // --- stage A tile: BM x BK fp32 -> bf16 LDS. 1024 float4, 4/thread
            #pragma unroll
            for (int i = 0; i < 4; ++i) {
                int idx = tid + i * 256;
                int row = idx >> 3;       // 8 float4 per 32-wide row
                int c4  = idx & 7;
                const float4 v = *(const float4*)(Ag + (blockM + row) * (long long)K + kg + c4 * 4);
                uint2 p;
                p.x = pack2bf(v.x, v.y);
                p.y = pack2bf(v.z, v.w);
                *(uint2*)&As[row][c4 * 4] = p;
            }
            // --- stage W tile transposed: [kg..kg+31][n0..n0+63] -> Bt[n][k]. 512 float4, 2/thread
            #pragma unroll
            for (int i = 0; i < 2; ++i) {
                int idx = tid + i * 256;
                int kr  = idx >> 4;       // 16 float4 per 64-wide k-row
                int c4  = idx & 15;
                const float4 v = *(const float4*)(Wg + (long long)(kg + kr) * N + blockN + c4 * 4);
                int n = c4 * 4;
                Bt[n + 0][kr] = f2bf(v.x);
                Bt[n + 1][kr] = f2bf(v.y);
                Bt[n + 2][kr] = f2bf(v.z);
                Bt[n + 3][kr] = f2bf(v.w);
            }
            __syncthreads();

            // --- fragments (ISA 16-bit layouts)
            // A: lanes 0-15 hold K {0..7,16..23} of row m; lanes 16-31 hold K {8..15,24..31}
            v16bf a0, a1, bf0, bf1;
            {
                const int m0 = wm * 32 + lm;
                const int kh = lhi * 8;
                ((uint4*)&a0)[0] = *(const uint4*)&As[m0][kh];
                ((uint4*)&a0)[1] = *(const uint4*)&As[m0][kh + 16];
                ((uint4*)&a1)[0] = *(const uint4*)&As[m0 + 16][kh];
                ((uint4*)&a1)[1] = *(const uint4*)&As[m0 + 16][kh + 16];
                // B: lanes 0-15 hold K {0..7,16..23} of col n; lanes 16-31 hold K {8..15,24..31}
                const int n0 = wn * 32 + lm;
                const int kl = lhi * 8;
                ((uint4*)&bf0)[0] = *(const uint4*)&Bt[n0][kl];
                ((uint4*)&bf0)[1] = *(const uint4*)&Bt[n0][kl + 16];
                ((uint4*)&bf1)[0] = *(const uint4*)&Bt[n0 + 16][kl];
                ((uint4*)&bf1)[1] = *(const uint4*)&Bt[n0 + 16][kl + 16];
            }
            acc00 = __builtin_amdgcn_wmma_f32_16x16x32_bf16(false, a0, false, bf0, (short)0, acc00, false, false);
            acc01 = __builtin_amdgcn_wmma_f32_16x16x32_bf16(false, a0, false, bf1, (short)0, acc01, false, false);
            acc10 = __builtin_amdgcn_wmma_f32_16x16x32_bf16(false, a1, false, bf0, (short)0, acc10, false, false);
            acc11 = __builtin_amdgcn_wmma_f32_16x16x32_bf16(false, a1, false, bf1, (short)0, acc11, false, false);
        }
    }

    // --- epilogue: leak + tanh.  C/D layout: VGPR r, lane L -> m = r + 8*(L>=16), n = L%16
    const float itau = 1.0f / args.tau;
    const float keep = 1.0f - itau;
    const int nl0 = wn * 32 + lm;
    const int nl1 = nl0 + 16;
    const float bA = bsum[nl0];
    const float bB = bsum[nl1];
    const long long ng0 = blockN + nl0;
    const long long ng1 = blockN + nl1;
    float* out = args.out;
    const float* prev = args.prev;

    #pragma unroll
    for (int r = 0; r < 8; ++r) {
        long long m  = blockM + wm * 32 + r + lhi * 8;
        long long m2 = m + 16;
        float s00 = (acc00[r] + bA) * itau;
        float s01 = (acc01[r] + bB) * itau;
        float s10 = (acc10[r] + bA) * itau;
        float s11 = (acc11[r] + bB) * itau;
        if (prev) {
            s00 += keep * prev[m  * N + ng0];
            s01 += keep * prev[m  * N + ng1];
            s10 += keep * prev[m2 * N + ng0];
            s11 += keep * prev[m2 * N + ng1];
        }
        out[m  * N + ng0] = tanhf(s00);
        out[m  * N + ng1] = tanhf(s01);
        out[m2 * N + ng0] = tanhf(s10);
        out[m2 * N + ng1] = tanhf(s11);
    }
}

// One launch for all four fused GEMM problems. blockIdx.x enumerates the
// concatenated N-tiles of (new_fast | new_slow | new_io | output); the branch
// is uniform per block -> scalar control flow, single LDS allocation.
__global__ __launch_bounds__(256)
void mtrnn_fused_all(FusedArgs pa, FusedArgs pb, FusedArgs pc, FusedArgs pd,
                     int s1, int s2, int s3) {
    const int bx = blockIdx.x;
    if (bx < s1) {
        run_problem(pa, bx);
    } else if (bx < s2) {
        run_problem(pb, bx - s1);
    } else if (bx < s3) {
        run_problem(pc, bx - s2);
    } else {
        run_problem(pd, bx - s3);
    }
}

extern "C" void kernel_launch(void* const* d_in, const int* in_sizes, int n_in,
                              void* d_out, int out_size, void* d_ws, size_t ws_size,
                              hipStream_t stream) {
    (void)n_in; (void)out_size; (void)d_ws; (void)ws_size;

    const float* input  = (const float*)d_in[0];
    const float* io_h   = (const float*)d_in[1];
    const float* fast_h = (const float*)d_in[2];
    const float* slow_h = (const float*)d_in[3];
    const float* W_in_io   = (const float*)d_in[4];  const float* b_in_io   = (const float*)d_in[5];
    const float* W_io_fast = (const float*)d_in[6];  const float* b_io_fast = (const float*)d_in[7];
    const float* W_ff      = (const float*)d_in[8];  const float* b_ff      = (const float*)d_in[9];
    const float* W_fs      = (const float*)d_in[10]; const float* b_fs      = (const float*)d_in[11];
    const float* W_ss      = (const float*)d_in[12]; const float* b_ss      = (const float*)d_in[13];
    const float* W_sf      = (const float*)d_in[14]; const float* b_sf      = (const float*)d_in[15];
    const float* W_fio     = (const float*)d_in[16]; const float* b_fio     = (const float*)d_in[17];
    const float* W_ioio    = (const float*)d_in[18]; const float* b_ioio    = (const float*)d_in[19];
    const float* W_ioout   = (const float*)d_in[20]; const float* b_ioout   = (const float*)d_in[21];

    const int IO = 64, IOH = 512, FH = 512, SH = 256;
    const long long Bsz = in_sizes[0] / IO;   // 16384

    float* out_output = (float*)d_out;                 // (B, 64)
    float* out_io     = out_output + Bsz * IO;         // (B, 512)
    float* out_fast   = out_io     + Bsz * IOH;        // (B, 512)
    float* out_slow   = out_fast   + Bsz * FH;         // (B, 256)

    // new_fast = leak(fast_h, fast_h@W_ff + slow_h@W_sf + io_h@W_io_fast + biases, tau=5); tanh
    FusedArgs pfast{ fast_h, W_ff, b_ff, FH,
                     slow_h, W_sf, b_sf, SH,
                     io_h,   W_io_fast, b_io_fast, IOH,
                     3, fast_h, out_fast, FH, 5.0f };
    // new_slow = leak(slow_h, slow_h@W_ss + fast_h@W_fs + biases, tau=70); tanh
    FusedArgs pslow{ slow_h, W_ss, b_ss, SH,
                     fast_h, W_fs, b_fs, FH,
                     fast_h, W_fs, b_fs, FH,   // unused (nseg=2)
                     2, slow_h, out_slow, SH, 70.0f };
    // new_io = leak(io_h, io_h@W_ioio + fast_h@W_fio + input@W_in_io + biases, tau=2); tanh
    FusedArgs pio  { io_h,   W_ioio,  b_ioio,  IOH,
                     fast_h, W_fio,   b_fio,   FH,
                     input,  W_in_io, b_in_io, IO,
                     3, io_h, out_io, IOH, 2.0f };
    // output = tanh(io_h @ W_io_out + b_io_out)   (tau=1 -> keep=0, prev=nullptr)
    FusedArgs pout { io_h, W_ioout, b_ioout, IOH,
                     io_h, W_ioout, b_ioout, IOH,  // unused
                     io_h, W_ioout, b_ioout, IOH,  // unused
                     1, nullptr, out_output, IO, 1.0f };

    const int ntFast = FH / BN;    // 8
    const int ntSlow = SH / BN;    // 4
    const int ntIo   = IOH / BN;   // 8
    const int ntOut  = IO / BN;    // 1
    const int s1 = ntFast;                 // 8
    const int s2 = s1 + ntSlow;            // 12
    const int s3 = s2 + ntIo;              // 20
    const int sx = s3 + ntOut;             // 21

    const unsigned mTiles = (unsigned)(Bsz / BM);   // 128
    mtrnn_fused_all<<<dim3((unsigned)sx, mTiles), dim3(256), 0, stream>>>(
        pfast, pslow, pio, pout, s1, s2, s3);
}
